// AdjacencyLearn_42734924595616
// MI455X (gfx1250) — compile-verified, matrix operand order checked by reading, not run on hardware
//
#include <hip/hip_runtime.h>
#include <math.h>

// ---------------------------------------------------------------------------
// NRI forward for MI455X (gfx1250). All H=128 GEMMs run on v_wmma_f32_16x16x32_f16
// (f16 A/B, f32 accumulate). Small-K / small-N linears stay on VALU.
// ---------------------------------------------------------------------------

typedef _Float16 f16;
typedef __attribute__((ext_vector_type(16))) _Float16 v16h;
typedef __attribute__((ext_vector_type(8)))  _Float16 v8h;
typedef __attribute__((ext_vector_type(8)))  float    v8f;

#define Bc 16
#define Nn 40
#define Tt 20
#define Dd 4
#define Hh 128
#define Ee 1560
#define MN (Bc*Nn)      // 640 node rows
#define ME (Bc*Ee)      // 24960 edge rows
#define TAUc 0.5f
#define GEPSc 1e-10f
#define BNEPS 1e-5f

// activation codes
#define ACT_NONE 0
#define ACT_ELU  1
#define ACT_RELU 2
#define ACT_TANH 3

__device__ __forceinline__ float act_apply(float x, int act) {
  switch (act) {
    case ACT_ELU:  return x > 0.f ? x : (__expf(x) - 1.f);
    case ACT_RELU: return x > 0.f ? x : 0.f;
    case ACT_TANH: return tanhf(x);
    default:       return x;
  }
}
__device__ __forceinline__ float sigm(float x) { return 1.f / (1.f + __expf(-x)); }

// Load a 16x32 f16 fragment (row-major source, ld elements) per ISA 7.12.2:
// lane = row (mod 16); lanes 0-15 hold K 0..7 / 16..23, lanes 16-31 hold 8..15 / 24..31.
// Two contiguous 16B loads per lane.
__device__ __forceinline__ v16h ld_frag(const f16* base) {
  v8h lo = *(const v8h*)(base);
  v8h hi = *(const v8h*)(base + 16);
  v16h r;
#pragma unroll
  for (int j = 0; j < 8; ++j) { r[j] = lo[j]; r[j + 8] = hi[j]; }
  return r;
}

// C = act(A[M,Kp] @ W[N=128,Kp]^T + bias) (* rowscale). One wave per 16x16 tile;
// 8 waves/block cover the 128-wide output. Kp multiple of 32. ldc = 128.
__global__ __launch_bounds__(256) void gemm_wmma(
    const f16* __restrict__ A, int lda, const f16* __restrict__ W, int Kp,
    const float* __restrict__ bias, int act, const float* __restrict__ rowscale,
    float* __restrict__ outF, f16* __restrict__ outH, int M) {
  const int wave = threadIdx.x >> 5;        // n-tile 0..7
  const int lane = threadIdx.x & 31;
  const int row  = lane & 15;
  const int hsel = lane >> 4;
  const int m0 = blockIdx.x * 16;
  const int n0 = wave * 16;

  const f16* arow = A + (size_t)(m0 + row) * lda + hsel * 8;
  const f16* wrow = W + (size_t)(n0 + row) * Kp + hsel * 8;

  v8f acc = {};
  for (int kk = 0; kk < Kp; kk += 32) {
    v16h a = ld_frag(arow + kk);
    v16h b = ld_frag(wrow + kk);
    if (kk + 32 < Kp) __builtin_prefetch(arow + kk + 32, 0, 1);
    acc = __builtin_amdgcn_wmma_f32_16x16x32_f16(
        false, a, false, b, (short)0, acc, false, false);
  }

  const int gn = n0 + (lane & 15);
  const float bv = bias ? bias[gn] : 0.f;
#pragma unroll
  for (int r = 0; r < 8; ++r) {
    const int gm = m0 + hsel * 8 + r;           // C/D layout: VGPR r -> M=r (+8 hi lanes)
    float v = act_apply(acc[r] + bv, act);
    if (rowscale) v *= rowscale[gm];
    if (outF) outF[(size_t)gm * Hh + gn] = v;
    if (outH) outH[(size_t)gm * Hh + gn] = (f16)v;
  }
}

// f32 [rows,Kd] -> f16 [rows,Kp] with zero pad (weights / first-layer input)
__global__ void convpad_f16(const float* __restrict__ s, f16* __restrict__ d,
                            int rows, int Kd, int Kp) {
  int i = blockIdx.x * blockDim.x + threadIdx.x;
  if (i >= rows * Kp) return;
  int r = i / Kp, k = i - r * Kp;
  d[i] = (f16)(k < Kd ? s[r * Kd + k] : 0.f);
}

__global__ void conv_f16s(const float* __restrict__ s, f16* __restrict__ d,
                          int n, float scale) {
  int i = blockIdx.x * blockDim.x + threadIdx.x;
  if (i < n) d[i] = (f16)(s[i] * scale);
}

__global__ void fill0(float* __restrict__ p, int n) {
  int i = blockIdx.x * blockDim.x + threadIdx.x;
  if (i < n) p[i] = 0.f;
}

// node2edge gather: out[b,e,:H]=x[b,rec[e]], out[b,e,H:2H]=x[b,send[e]]  (f16, row stride ld)
__global__ void gather2(const float* __restrict__ x, const int* __restrict__ rec,
                        const int* __restrict__ send, f16* __restrict__ out, int ld) {
  int i = blockIdx.x * blockDim.x + threadIdx.x;
  if (i >= ME * 2 * Hh) return;
  int col = i % (2 * Hh);
  int r = i / (2 * Hh);
  int e = r % Ee, b = r / Ee;
  float v = (col < Hh) ? x[((size_t)b * Nn + rec[e]) * Hh + col]
                       : x[((size_t)b * Nn + send[e]) * Hh + col - Hh];
  out[(size_t)r * ld + col] = (f16)v;
}

// [x[rec], x[send], skip] -> f16 [ME,384]
__global__ void gather3(const float* __restrict__ x, const float* __restrict__ skip,
                        const int* __restrict__ rec, const int* __restrict__ send,
                        f16* __restrict__ out) {
  int i = blockIdx.x * blockDim.x + threadIdx.x;
  if (i >= ME * 3 * Hh) return;
  int col = i % (3 * Hh);
  int r = i / (3 * Hh);
  int e = r % Ee, b = r / Ee;
  float v;
  if (col < Hh)            v = x[((size_t)b * Nn + rec[e]) * Hh + col];
  else if (col < 2 * Hh)   v = x[((size_t)b * Nn + send[e]) * Hh + col - Hh];
  else                     v = skip[(size_t)r * Hh + col - 2 * Hh];
  out[(size_t)r * 384 + col] = (f16)v;
}

// edge2node scatter-add (f32 atomics -> global_atomic_add_f32)
__global__ void scatter_add(const float* __restrict__ src, const int* __restrict__ rec,
                            float* __restrict__ dst) {
  int i = blockIdx.x * blockDim.x + threadIdx.x;
  if (i >= ME * Hh) return;
  int h = i % Hh;
  int r = i / Hh;
  int e = r % Ee, b = r / Ee;
  atomicAdd(&dst[((size_t)b * Nn + rec[e]) * Hh + h], src[i]);
}

// BN stats: sum & sumsq per feature (128 features). 32768 threads, thread owns feature i%128.
__global__ void bn_stats(const float* __restrict__ x, int M, float* __restrict__ st) {
  int g = blockIdx.x * blockDim.x + threadIdx.x;   // 128 blocks * 256
  int f = g & 127;
  float s = 0.f, ss = 0.f;
  for (int r = g >> 7; r < M; r += 256) {
    float v = x[(size_t)r * Hh + f];
    s += v; ss += v * v;
  }
  atomicAdd(&st[f], s);
  atomicAdd(&st[128 + f], ss);
}

__global__ void bn_apply(const float* __restrict__ x, float* __restrict__ y,
                         const float* __restrict__ st, const float* __restrict__ g,
                         const float* __restrict__ b, int M) {
  int i = blockIdx.x * blockDim.x + threadIdx.x;
  if (i >= M * Hh) return;
  int f = i % Hh;
  float m = st[f] / (float)M;
  float v = st[128 + f] / (float)M - m * m;
  y[i] = (x[i] - m) * rsqrtf(v + BNEPS) * g[f] + b[f];
}

// fc_out (K=2) + gumbel softmax -> rel_type[:, :, 1]
__global__ void gumbel_rel(const float* __restrict__ x, const float* __restrict__ fw,
                           const float* __restrict__ fb, const float* __restrict__ u,
                           float* __restrict__ rel1) {
  int r = blockIdx.x * blockDim.x + threadIdx.x;
  if (r >= ME) return;
  float l0 = fb[0], l1 = fb[1];
  const float* xr = x + (size_t)r * Hh;
  for (int k = 0; k < Hh; ++k) { l0 += xr[k] * fw[k]; l1 += xr[k] * fw[Hh + k]; }
  float g0 = -logf(GEPSc - logf(u[2 * r]     + GEPSc));
  float g1 = -logf(GEPSc - logf(u[2 * r + 1] + GEPSc));
  float y = ((l0 + g0) - (l1 + g1)) / TAUc;
  rel1[r] = 1.f / (1.f + __expf(y));
}

// generic small linear: out[M,Nc] = A[M,(lda)] @ W[Nc,Kd]^T + bias
__global__ void lin_small(const float* __restrict__ A, int lda, const float* __restrict__ W,
                          const float* __restrict__ bias, float* __restrict__ out,
                          int M, int Nc, int Kd) {
  int i = blockIdx.x * blockDim.x + threadIdx.x;
  if (i >= M * Nc) return;
  int row = i / Nc, n = i - row * Nc;
  float acc = bias ? bias[n] : 0.f;
  for (int k = 0; k < Kd; ++k) acc += A[(size_t)row * lda + k] * W[n * Kd + k];
  out[i] = acc;
}

// GRU pointwise: r=sig(ar+hr) i=sig(ai+hi) n=tanh(an+r*hn) h=(1-i)*n+i*h
__global__ void gru_pw(const float* ar, const float* ai, const float* an,
                       const float* hr, const float* hi, const float* hn,
                       float* __restrict__ h) {
  int i = blockIdx.x * blockDim.x + threadIdx.x;
  if (i >= MN * Hh) return;
  float r = sigm(ar[i] + hr[i]);
  float ii = sigm(ai[i] + hi[i]);
  float nn = tanhf(an[i] + r * hn[i]);
  h[i] = (1.f - ii) * nn + ii * h[i];
}

// pred = ins + pr @ of3^T + b, stored at out[b,n,t,:]
__global__ void pred_out(const float* __restrict__ pr, const float* __restrict__ w,
                         const float* __restrict__ b, const float* __restrict__ data,
                         int t, float* __restrict__ out) {
  int i = blockIdx.x * blockDim.x + threadIdx.x;
  if (i >= MN * Dd) return;
  int row = i >> 2, d = i & 3;
  float acc = b[d];
  const float* p = pr + (size_t)row * Hh;
  for (int k = 0; k < Hh; ++k) acc += p[k] * w[d * Hh + k];
  float ins = data[(size_t)row * (Tt * Dd) + t * Dd + d];
  out[(size_t)row * ((Tt - 1) * Dd) + t * Dd + d] = ins + acc;
}

// ---------------------------------------------------------------------------
static inline int gsz(int n) { return (n + 255) / 256; }

extern "C" void kernel_launch(void* const* d_in, const int* in_sizes, int n_in,
                              void* d_out, int out_size, void* d_ws, size_t ws_size,
                              hipStream_t stream) {
  // Inputs flattened jax-tree style (dict keys sorted):
  // data, gumbel_u, params{fc_out_b, fc_out_w, hi_w, hn_w, hr_w, ii_b, ii_w,
  //   in_b, in_w, ir_b, ir_w, mlp1{b1,b2,bb,g,w1,w2}, mlp2{..}, mlp3{..}, mlp4{..},
  //   msg1_b, msg1_w, msg2_b, msg2_w, of1_b, of1_w, of2_b, of2_w, of3_b, of3_w},
  // rec_idx, send_idx
  const float* data    = (const float*)d_in[0];
  const float* gumb    = (const float*)d_in[1];
  const float* fc_b    = (const float*)d_in[2];
  const float* fc_w    = (const float*)d_in[3];
  const float* hi_w    = (const float*)d_in[4];
  const float* hn_w    = (const float*)d_in[5];
  const float* hr_w    = (const float*)d_in[6];
  const float* ii_b    = (const float*)d_in[7];
  const float* ii_w    = (const float*)d_in[8];
  const float* in_b    = (const float*)d_in[9];
  const float* in_w    = (const float*)d_in[10];
  const float* ir_b    = (const float*)d_in[11];
  const float* ir_w    = (const float*)d_in[12];
  const float* m_b1[4], *m_b2[4], *m_bb[4], *m_g[4], *m_w1[4], *m_w2[4];
  for (int m = 0; m < 4; ++m) {
    int base = 13 + 6 * m;
    m_b1[m] = (const float*)d_in[base + 0];
    m_b2[m] = (const float*)d_in[base + 1];
    m_bb[m] = (const float*)d_in[base + 2];
    m_g[m]  = (const float*)d_in[base + 3];
    m_w1[m] = (const float*)d_in[base + 4];
    m_w2[m] = (const float*)d_in[base + 5];
  }
  const float* msg1_b = (const float*)d_in[37];
  const float* msg1_w = (const float*)d_in[38];
  const float* msg2_b = (const float*)d_in[39];
  const float* msg2_w = (const float*)d_in[40];
  const float* of1_b  = (const float*)d_in[41];
  const float* of1_w  = (const float*)d_in[42];
  const float* of2_b  = (const float*)d_in[43];
  const float* of2_w  = (const float*)d_in[44];
  const float* of3_b  = (const float*)d_in[45];
  const float* of3_w  = (const float*)d_in[46];
  const int* rec  = (const int*)d_in[47];
  const int* send = (const int*)d_in[48];
  float* out = (float*)d_out;

  // ---- workspace arena ----
  char* ws = (char*)d_ws;
  size_t off = 0;
  auto alloc = [&](size_t bytes) -> void* {
    void* p = ws + off; off = (off + bytes + 255) & ~(size_t)255; return p;
  };
  // f16 weights (K padded to mult of 32)
  f16* w_m1_1 = (f16*)alloc(128 * 96 * 2);
  f16* w_m1_2 = (f16*)alloc(128 * 128 * 2);
  f16* w_m2_1 = (f16*)alloc(128 * 256 * 2);
  f16* w_m2_2 = (f16*)alloc(128 * 128 * 2);
  f16* w_m3_1 = (f16*)alloc(128 * 128 * 2);
  f16* w_m3_2 = (f16*)alloc(128 * 128 * 2);
  f16* w_m4_1 = (f16*)alloc(128 * 384 * 2);
  f16* w_m4_2 = (f16*)alloc(128 * 128 * 2);
  f16* w_msg1 = (f16*)alloc(128 * 256 * 2);
  f16* w_msg2 = (f16*)alloc(128 * 128 * 2);
  f16* w_hr   = (f16*)alloc(128 * 128 * 2);
  f16* w_hi   = (f16*)alloc(128 * 128 * 2);
  f16* w_hn   = (f16*)alloc(128 * 128 * 2);
  f16* w_of1  = (f16*)alloc(128 * 128 * 2);
  f16* w_of2  = (f16*)alloc(128 * 128 * 2);
  // activations
  f16*   x_in16 = (f16*)alloc((size_t)MN * 96 * 2);
  f16*   n16a   = (f16*)alloc((size_t)MN * Hh * 2);
  f16*   n16b   = (f16*)alloc((size_t)MN * Hh * 2);
  f16*   agg16  = (f16*)alloc((size_t)MN * Hh * 2);
  f16*   h16    = (f16*)alloc((size_t)MN * Hh * 2);
  float* nA     = (float*)alloc((size_t)MN * Hh * 4);
  float* nB     = (float*)alloc((size_t)MN * Hh * 4);
  float* ga_r   = (float*)alloc((size_t)MN * Hh * 4);
  float* ga_i   = (float*)alloc((size_t)MN * Hh * 4);
  float* ga_n   = (float*)alloc((size_t)MN * Hh * 4);
  float* gh_r   = (float*)alloc((size_t)MN * Hh * 4);
  float* gh_i   = (float*)alloc((size_t)MN * Hh * 4);
  float* gh_n   = (float*)alloc((size_t)MN * Hh * 4);
  float* agg    = (float*)alloc((size_t)MN * Hh * 4);
  float* hidden = (float*)alloc((size_t)MN * Hh * 4);
  float* stats  = (float*)alloc(256 * 4);
  f16*   eIn16  = (f16*)alloc((size_t)ME * 384 * 2);
  f16*   e16    = (f16*)alloc((size_t)ME * Hh * 2);
  float* eA     = (float*)alloc((size_t)ME * Hh * 4);
  float* eSkip  = (float*)alloc((size_t)ME * Hh * 4);
  float* rel1   = (float*)alloc((size_t)ME * 4);
  (void)ws_size; (void)n_in; (void)in_sizes; (void)out_size;

  auto CP = [&](const float* s, f16* d, int rows, int Kd, int Kp) {
    convpad_f16<<<gsz(rows * Kp), 256, 0, stream>>>(s, d, rows, Kd, Kp);
  };
  auto GEMM = [&](const f16* A, int lda, const f16* W, int Kp, const float* bias,
                  int act, const float* rs, float* oF, f16* oH, int M) {
    gemm_wmma<<<dim3(M / 16), dim3(256), 0, stream>>>(A, lda, W, Kp, bias, act, rs, oF, oH, M);
  };
  auto ZERO = [&](float* p, int n) { fill0<<<gsz(n), 256, 0, stream>>>(p, n); };
  auto BN = [&](const float* x, float* y, const float* g, const float* b, int M) {
    ZERO(stats, 256);
    bn_stats<<<128, 256, 0, stream>>>(x, M, stats);
    bn_apply<<<gsz(M * Hh), 256, 0, stream>>>(x, y, stats, g, b, M);
  };

  // ---- weight conversion (f32 -> padded f16) ----
  CP(m_w1[0], w_m1_1, 128, 80, 96);   CP(m_w2[0], w_m1_2, 128, 128, 128);
  CP(m_w1[1], w_m2_1, 128, 256, 256); CP(m_w2[1], w_m2_2, 128, 128, 128);
  CP(m_w1[2], w_m3_1, 128, 128, 128); CP(m_w2[2], w_m3_2, 128, 128, 128);
  CP(m_w1[3], w_m4_1, 128, 384, 384); CP(m_w2[3], w_m4_2, 128, 128, 128);
  CP(msg1_w + 128 * 256, w_msg1, 128, 256, 256);  // edge type k=1 (skip_first)
  CP(msg2_w + 128 * 128, w_msg2, 128, 128, 128);
  CP(hr_w, w_hr, 128, 128, 128); CP(hi_w, w_hi, 128, 128, 128); CP(hn_w, w_hn, 128, 128, 128);
  CP(of1_w, w_of1, 128, 128, 128); CP(of2_w, w_of2, 128, 128, 128);

  // ---- encoder ----
  CP(data, x_in16, MN, 80, 96);                                        // [640,80]->[640,96]
  GEMM(x_in16, 96, w_m1_1, 96, m_b1[0], ACT_ELU, nullptr, nullptr, n16a, MN);
  GEMM(n16a, 128, w_m1_2, 128, m_b2[0], ACT_ELU, nullptr, nA, nullptr, MN);
  BN(nA, nB, m_g[0], m_bb[0], MN);                                     // x nodes
  gather2<<<gsz(ME * 2 * Hh), 256, 0, stream>>>(nB, rec, send, eIn16, 256);
  GEMM(eIn16, 256, w_m2_1, 256, m_b1[1], ACT_ELU, nullptr, nullptr, e16, ME);
  GEMM(e16, 128, w_m2_2, 128, m_b2[1], ACT_ELU, nullptr, eA, nullptr, ME);
  BN(eA, eSkip, m_g[1], m_bb[1], ME);                                  // x_skip
  ZERO(nA, MN * Hh);
  scatter_add<<<gsz(ME * Hh), 256, 0, stream>>>(eSkip, rec, nA);
  conv_f16s<<<gsz(MN * Hh), 256, 0, stream>>>(nA, n16a, MN * Hh, 1.f / Nn);
  GEMM(n16a, 128, w_m3_1, 128, m_b1[2], ACT_ELU, nullptr, nullptr, n16b, MN);
  GEMM(n16b, 128, w_m3_2, 128, m_b2[2], ACT_ELU, nullptr, nA, nullptr, MN);
  BN(nA, nB, m_g[2], m_bb[2], MN);
  gather3<<<gsz(ME * 3 * Hh), 256, 0, stream>>>(nB, eSkip, rec, send, eIn16);
  GEMM(eIn16, 384, w_m4_1, 384, m_b1[3], ACT_ELU, nullptr, nullptr, e16, ME);
  GEMM(e16, 128, w_m4_2, 128, m_b2[3], ACT_ELU, nullptr, eA, nullptr, ME);
  BN(eA, eSkip, m_g[3], m_bb[3], ME);                                  // final edge feats
  gumbel_rel<<<gsz(ME), 256, 0, stream>>>(eSkip, fc_w, fc_b, gumb, rel1);

  // ---- recurrent decoder ----
  ZERO(hidden, MN * Hh);
  for (int t = 0; t < Tt - 1; ++t) {
    gather2<<<gsz(ME * 2 * Hh), 256, 0, stream>>>(hidden, rec, send, eIn16, 256);
    GEMM(eIn16, 256, w_msg1, 256, msg1_b + Hh, ACT_TANH, nullptr, nullptr, e16, ME);
    GEMM(e16, 128, w_msg2, 128, msg2_b + Hh, ACT_TANH, rel1, eA, nullptr, ME);
    ZERO(agg, MN * Hh);
    scatter_add<<<gsz(ME * Hh), 256, 0, stream>>>(eA, rec, agg);
    conv_f16s<<<gsz(MN * Hh), 256, 0, stream>>>(agg, agg16, MN * Hh, 1.f / Dd);
    lin_small<<<gsz(MN * Hh), 256, 0, stream>>>(data + t * Dd, Tt * Dd, ir_w, ir_b, ga_r, MN, Hh, Dd);
    lin_small<<<gsz(MN * Hh), 256, 0, stream>>>(data + t * Dd, Tt * Dd, ii_w, ii_b, ga_i, MN, Hh, Dd);
    lin_small<<<gsz(MN * Hh), 256, 0, stream>>>(data + t * Dd, Tt * Dd, in_w, in_b, ga_n, MN, Hh, Dd);
    GEMM(agg16, 128, w_hr, 128, nullptr, ACT_NONE, nullptr, gh_r, nullptr, MN);
    GEMM(agg16, 128, w_hi, 128, nullptr, ACT_NONE, nullptr, gh_i, nullptr, MN);
    GEMM(agg16, 128, w_hn, 128, nullptr, ACT_NONE, nullptr, gh_n, nullptr, MN);
    gru_pw<<<gsz(MN * Hh), 256, 0, stream>>>(ga_r, ga_i, ga_n, gh_r, gh_i, gh_n, hidden);
    conv_f16s<<<gsz(MN * Hh), 256, 0, stream>>>(hidden, h16, MN * Hh, 1.f);
    GEMM(h16, 128, w_of1, 128, of1_b, ACT_RELU, nullptr, nullptr, n16a, MN);
    GEMM(n16a, 128, w_of2, 128, of2_b, ACT_RELU, nullptr, nA, nullptr, MN);
    pred_out<<<gsz(MN * Dd), 256, 0, stream>>>(nA, of3_w, of3_b, data, t, out);
  }
}
